// SelfAttentionLayer_20461224198666
// MI455X (gfx1250) — matrix operand order repaired and checked by reading, not compiled, not run on hardware
//
#include <hip/hip_runtime.h>

// ---------------------------------------------------------------------------
// Self-attention (B=8, T=1024, D_IN=512, D_H=64, N_H=8) for gfx1250 (MI455X).
// bf16 WMMA (v_wmma_f32_16x16x32_bf16) for all matmuls, f32 accumulation,
// flash-style online softmax so the (B,T,T,H) score tensor is never stored.
// ---------------------------------------------------------------------------

typedef __attribute__((ext_vector_type(16))) __bf16        v16bf;
typedef __attribute__((ext_vector_type(8)))  float         v8f;
typedef __attribute__((ext_vector_type(8)))  unsigned int  v8u;

constexpr int B_   = 8;
constexpr int T_   = 1024;
constexpr int DIN  = 512;
constexpr int DH   = 64;
constexpr int NH   = 8;

static __device__ __forceinline__ v8f wmma_bf16(v8u a, v8u b, v8f c) {
  return __builtin_amdgcn_wmma_f32_16x16x32_bf16(
      /*neg_a=*/false, __builtin_bit_cast(v16bf, a),
      /*neg_b=*/false, __builtin_bit_cast(v16bf, b),
      /*c_mod=*/(short)0, c, /*reuse_a=*/false, /*reuse_b=*/false);
}

static __device__ __forceinline__ unsigned int pack_bf16(float x, float y) {
  union { __bf16 b[2]; unsigned int u; } t;
  t.b[0] = (__bf16)x; t.b[1] = (__bf16)y;
  return t.u;
}

// A-operand (16x32 bf16) k-index for VGPR-pair v (0..7) and lane-half:
//   k = 2*v + 16*half
static __device__ __forceinline__ int a_k(int v, int half) {
  return 2 * v + 16 * half;
}

// ---------------------------------------------------------------------------
// Kernel 1: W (i,d,h) f32  ->  Wt [which][h][d][i] bf16 (K-pairs contiguous)
// ---------------------------------------------------------------------------
__global__ void wconv_kernel(const float* __restrict__ Wk,
                             const float* __restrict__ Wq,
                             const float* __restrict__ Wv,
                             __bf16* __restrict__ wt) {
  const int which = blockIdx.y;
  const float* W = (which == 0) ? Wk : ((which == 1) ? Wq : Wv);
  __bf16* dst = wt + (size_t)which * (NH * DH * DIN);
  const int n = NH * DH * DIN;
  for (int idx = blockIdx.x * blockDim.x + threadIdx.x; idx < n;
       idx += gridDim.x * blockDim.x) {
    const int i = idx % DIN;
    const int d = (idx / DIN) % DH;
    const int h = idx / (DIN * DH);
    dst[idx] = (__bf16)W[(i * DH + d) * NH + h];
  }
}

// ---------------------------------------------------------------------------
// Kernel 2: projections.  One wave computes a 16(t) x 64(d) tile per (b,h).
//   which 0: Kh[bh][t][d]   which 1: Qh[bh][t][d]   which 2: Vt[bh][d][t]
// ---------------------------------------------------------------------------
__global__ __launch_bounds__(256)
void proj_kernel(const float* __restrict__ keys,
                 const float* __restrict__ queries,
                 const float* __restrict__ values,
                 const __bf16* __restrict__ wt,
                 __bf16* __restrict__ Kh,
                 __bf16* __restrict__ Qh,
                 __bf16* __restrict__ Vt) {
  const int which = blockIdx.z;
  const float*  X = (which == 0) ? keys : ((which == 1) ? queries : values);
  const __bf16* W = wt + (size_t)which * (NH * DH * DIN);

  const int bh   = blockIdx.y;
  const int b    = bh >> 3;
  const int h    = bh & 7;
  const int wave = threadIdx.x >> 5;
  const int lane = threadIdx.x & 31;
  const int half = lane >> 4;
  const int l16  = lane & 15;
  const int t0   = (blockIdx.x * 8 + wave) * 16;

  const float* Xrow = X + ((size_t)b * T_ + (t0 + l16)) * DIN;

  v8f acc[4];
#pragma unroll
  for (int nt = 0; nt < 4; ++nt) acc[nt] = {};

  for (int kk = 0; kk < DIN; kk += 32) {
    v8u a;
#pragma unroll
    for (int v = 0; v < 8; ++v) {
      const int k = kk + a_k(v, half);
      a[v] = pack_bf16(Xrow[k], Xrow[k + 1]);
    }
#pragma unroll
    for (int nt = 0; nt < 4; ++nt) {
      const int d = nt * 16 + l16;
      const __bf16* wp = W + ((size_t)(h * DH + d)) * DIN + kk + 16 * half;
      v8u bf;
#pragma unroll
      for (int v = 0; v < 8; ++v)
        bf[v] = *(const unsigned int*)(wp + 2 * v);
      acc[nt] = wmma_bf16(a, bf, acc[nt]);
    }
  }

  if (which != 2) {
    __bf16* dst = ((which == 0) ? Kh : Qh) + (size_t)bh * (T_ * DH);
#pragma unroll
    for (int nt = 0; nt < 4; ++nt)
#pragma unroll
      for (int r = 0; r < 8; ++r) {
        const int t = t0 + r + 8 * half;
        const int d = nt * 16 + l16;
        dst[(size_t)t * DH + d] = (__bf16)acc[nt][r];
      }
  } else {
    __bf16* dst = Vt + (size_t)bh * (DH * T_);
#pragma unroll
    for (int nt = 0; nt < 4; ++nt)
#pragma unroll
      for (int r = 0; r < 8; ++r) {
        const int t = t0 + r + 8 * half;
        const int d = nt * 16 + l16;
        dst[(size_t)d * T_ + t] = (__bf16)acc[nt][r];
      }
  }
}

// ---------------------------------------------------------------------------
// Kernel 3: fused flash attention. One wave: 16 queries, keys in chunks of 32.
// ---------------------------------------------------------------------------
__global__ __launch_bounds__(256)
void attn_kernel(const __bf16* __restrict__ Kh,
                 const __bf16* __restrict__ Qh,
                 const __bf16* __restrict__ Vt,
                 const int*    __restrict__ key_seq,
                 float*        __restrict__ out) {
  // Per-wave staging of P^T (16 j-rows x 32 i-cols), bf16, padded row stride.
  __shared__ __align__(16) __bf16 PtS[8][16][36];

  const int bh   = blockIdx.y;
  const int b    = bh >> 3;
  const int h    = bh & 7;
  const int wave = threadIdx.x >> 5;
  const int lane = threadIdx.x & 31;
  const int half = lane >> 4;
  const int l16  = lane & 15;
  const int j0   = (blockIdx.x * 8 + wave) * 16;
  const int j    = j0 + l16;   // this lane's query column

  const __bf16* Kb = Kh + (size_t)bh * (T_ * DH);
  const __bf16* Qb = Qh + (size_t)bh * (T_ * DH);
  const __bf16* Vb = Vt + (size_t)bh * (DH * T_);
  const int*    ks = key_seq + (size_t)b * T_;

  __bf16 (*Pt)[36] = PtS[wave];

  // Q as B-operand (k=d, n=j), two d-slices of 32. Hoisted out of the loop.
  v8u bq[2];
#pragma unroll
  for (int s = 0; s < 2; ++s) {
    const __bf16* qp = Qb + (size_t)j * DH + s * 32 + 16 * half;
#pragma unroll
    for (int v = 0; v < 8; ++v)
      bq[s][v] = *(const unsigned int*)(qp + 2 * v);
  }

  float m_run = -3.0e38f, l_run = 0.0f;
  v8f o[4];
#pragma unroll
  for (int nt = 0; nt < 4; ++nt) o[nt] = {};

  const int iend = j0 + 15;            // causal: keys beyond the tile are dead
  for (int i0 = 0; i0 <= iend; i0 += 32) {
    // ---- scores: S[g] = K[i0+16g .. +15, :] * Q^T  (16i x 16j, f32) ----
    v8f S[2];
#pragma unroll
    for (int g = 0; g < 2; ++g) {
      v8f s = {};
#pragma unroll
      for (int sl = 0; sl < 2; ++sl) {
        const __bf16* kp = Kb + (size_t)(i0 + g * 16 + l16) * DH + sl * 32;
        v8u a;
#pragma unroll
        for (int v = 0; v < 8; ++v)
          a[v] = *(const unsigned int*)(kp + a_k(v, half));
        s = wmma_bf16(a, bq[sl], s);
      }
      S[g] = s;
    }

    // ---- mask (before scaling, like the reference), scale by 1/sqrt(64) ----
    float sv[16];
    float cmax = -3.0e38f;
#pragma unroll
    for (int g = 0; g < 2; ++g)
#pragma unroll
      for (int r = 0; r < 8; ++r) {
        const int i = i0 + g * 16 + r + 8 * half;
        float x = S[g][r];
        float mask = 0.0f;
        if (ks[i] == -1) mask -= 1000.0f;  // pad mask on key positions
        if (i > j)       mask -= 1000.0f;  // causal mask
        x = (x + mask) * 0.125f;
        sv[g * 8 + r] = x;
        cmax = fmaxf(cmax, x);
      }
    cmax = fmaxf(cmax, __shfl_xor(cmax, 16));   // full 32-row column max

    const float m_new = fmaxf(m_run, cmax);
    const float alpha = __expf(m_run - m_new);

    float csum = 0.0f;
#pragma unroll
    for (int g = 0; g < 2; ++g)
#pragma unroll
      for (int r = 0; r < 8; ++r) {
        const float p = __expf(sv[g * 8 + r] - m_new);
        csum += p;
        Pt[l16][g * 16 + r + 8 * half] = (__bf16)p;   // P^T[j_local][i_local]
      }
    csum += __shfl_xor(csum, 16);
    l_run = l_run * alpha + csum;
    m_run = m_new;

    // ---- rescale O rows (row = j); alpha lives lane-indexed by j ----
#pragma unroll
    for (int r = 0; r < 8; ++r) {
      const float ar = __shfl(alpha, r + 8 * half);
#pragma unroll
      for (int nt = 0; nt < 4; ++nt) o[nt][r] *= ar;
    }

    // ---- P^T as A-operand (m=j_local, k=i_local) from LDS ----
    v8u ap;
#pragma unroll
    for (int v = 0; v < 8; ++v)
      ap[v] = *(const unsigned int*)(&Pt[l16][a_k(v, half)]);

    // ---- O += P^T * V  (4 WMMAs over d) ----
#pragma unroll
    for (int nt = 0; nt < 4; ++nt) {
      const int d = nt * 16 + l16;
      const __bf16* vp = Vb + (size_t)d * T_ + i0 + 16 * half;
      v8u bv;
#pragma unroll
      for (int v = 0; v < 8; ++v)
        bv[v] = *(const unsigned int*)(vp + 2 * v);
      o[nt] = wmma_bf16(ap, bv, o[nt]);
    }
  }

  // ---- normalize by l_j and store head-major: out[b][j][h*64 + d] ----
  const float linv = 1.0f / l_run;
#pragma unroll
  for (int r = 0; r < 8; ++r) {
    const float lr = __shfl(linv, r + 8 * half);
    const int jrow = j0 + r + 8 * half;
    float* orow = out + ((size_t)b * T_ + jrow) * (NH * DH) + h * DH;
#pragma unroll
    for (int nt = 0; nt < 4; ++nt)
      orow[nt * 16 + l16] = o[nt][r] * lr;
  }
}

// ---------------------------------------------------------------------------
// Launcher
// ---------------------------------------------------------------------------
extern "C" void kernel_launch(void* const* d_in, const int* in_sizes, int n_in,
                              void* d_out, int out_size, void* d_ws, size_t ws_size,
                              hipStream_t stream) {
  const float* keys    = (const float*)d_in[0];
  const float* queries = (const float*)d_in[1];
  const float* values  = (const float*)d_in[2];
  const float* Wk      = (const float*)d_in[3];
  const float* Wq      = (const float*)d_in[4];
  const float* Wv      = (const float*)d_in[5];
  const int*   key_seq = (const int*)d_in[6];
  float* out = (float*)d_out;

  // Workspace layout (bf16 elements):
  //   Wt : 3 * NH*DH*DIN              =   786,432
  //   Kh : B*NH*T*DH                  = 4,194,304
  //   Qh : B*NH*T*DH                  = 4,194,304
  //   Vt : B*NH*DH*T                  = 4,194,304   (~26.7 MB total)
  unsigned short* wsu = (unsigned short*)d_ws;
  __bf16* wt = (__bf16*)(wsu);
  __bf16* Kh = (__bf16*)(wsu + (size_t)3 * NH * DH * DIN);
  __bf16* Qh = Kh + (size_t)B_ * NH * T_ * DH;
  __bf16* Vt = Qh + (size_t)B_ * NH * T_ * DH;

  wconv_kernel<<<dim3(64, 3), 256, 0, stream>>>(Wk, Wq, Wv, wt);
  proj_kernel<<<dim3(8, B_ * NH, 3), 256, 0, stream>>>(keys, queries, values,
                                                       wt, Kh, Qh, Vt);
  attn_kernel<<<dim3(8, B_ * NH), 256, 0, stream>>>(Kh, Qh, Vt, key_seq, out);
}
